// self_attention_layer_39676907888369
// MI455X (gfx1250) — matrix operand order repaired and checked by reading, not compiled, not run on hardware
//
#include <hip/hip_runtime.h>
#include <math.h>

// ---------------- problem constants ----------------
#define B_    64
#define D_    1500
#define T_    1000
#define DA_   512
#define DPAD  1504          // 47 * 32 (K padded for 16x16x32 bf16 WMMA)
#define KT    47
#define TM    32            // t-rows per GEMM block

typedef __attribute__((ext_vector_type(16))) __bf16 v16bf;
typedef __attribute__((ext_vector_type(8)))  float  v8f;

union Frag { uint4 u[2]; v16bf v; };   // 32 bytes = 16 bf16 = one wmma operand/lane

// fp32 -> bf16 round-to-nearest-even (inputs are finite; no NaN path needed)
__device__ __forceinline__ unsigned short f2bf(float f) {
    union { float f; unsigned u; } c; c.f = f;
    unsigned r = c.u + 0x7FFFu + ((c.u >> 16) & 1u);
    return (unsigned short)(r >> 16);
}

// ---------------- kernel 1: W1 -> bf16, transposed [DA][DPAD], K zero-padded ----
__global__ void prep_w1(const float* __restrict__ w1, unsigned short* __restrict__ w1t) {
    int idx = blockIdx.x * 256 + threadIdx.x;          // over DA_*DPAD
    if (idx >= DA_ * DPAD) return;
    int n = idx / DPAD;
    int k = idx - n * DPAD;
    float v = (k < D_) ? w1[(size_t)k * DA_ + n] : 0.0f;
    w1t[idx] = f2bf(v);
}

// ---------------- kernel 2: scores[b][t] = w2 . relu(X^T W1 + b1) ----------------
// block = 256 threads (8 waves). Block tile: M=32 t-rows, full K=1504, full N=512.
// wave w owns N columns [w*64, w*64+64), 2 M-subtiles, 4 N-subtiles -> 8 WMMA/k-step.
__launch_bounds__(256)
__global__ void attn_scores(const float* __restrict__ x,          // [B,D,T]
                            const unsigned short* __restrict__ w1t, // [DA][DPAD] bf16
                            const float* __restrict__ b1,         // [DA]
                            const float* __restrict__ w2,         // [DA]
                            float* __restrict__ scores)           // [B,T]
{
    __shared__ __align__(16) unsigned short Aslab[TM][DPAD];      // 96,256 B (of 320 KB)
    __shared__ float sred[TM];

    const int b   = blockIdx.y;
    const int t0  = blockIdx.x * TM;
    const int tid = threadIdx.x;

    if (tid < TM) sred[tid] = 0.0f;

    // ---- stage X^T tile [TM x DPAD] into LDS as bf16 (fp32 global -> cvt -> ds) ----
    {
        const int tg = tid & 7;      // 8 threads cover the 32 t's (float4 each)
        const int dI = tid >> 3;     // 32 d's per sweep
        for (int d0 = 0; d0 < DPAD; d0 += 32) {
            const int d = d0 + dI;
            float4 v = make_float4(0.f, 0.f, 0.f, 0.f);
            const int t = t0 + tg * 4;
            if (d < D_) {
                const float* p = x + ((size_t)b * D_ + d) * T_ + t;
                if (t + 3 < T_) {
                    v = *(const float4*)p;
                } else {
                    if (t + 0 < T_) v.x = p[0];
                    if (t + 1 < T_) v.y = p[1];
                    if (t + 2 < T_) v.z = p[2];
                    if (t + 3 < T_) v.w = p[3];
                }
            }
            Aslab[tg * 4 + 0][d] = f2bf(v.x);
            Aslab[tg * 4 + 1][d] = f2bf(v.y);
            Aslab[tg * 4 + 2][d] = f2bf(v.z);
            Aslab[tg * 4 + 3][d] = f2bf(v.w);
        }
    }
    __syncthreads();

    const int wave  = tid >> 5;
    const int lane  = tid & 31;
    const int lhi   = lane >> 4;        // half-wave select
    const int lm    = lane & 15;
    const int nbase = wave * 64;

    const v8f vzero = {0.f, 0.f, 0.f, 0.f, 0.f, 0.f, 0.f, 0.f};
    v8f acc[2][4];
    #pragma unroll
    for (int m = 0; m < 2; ++m)
        #pragma unroll
        for (int n = 0; n < 4; ++n) acc[m][n] = vzero;

    // ---- K loop: 47 steps of K=32 ----
    for (int kt = 0; kt < KT; ++kt) {
        // A fragments (16-bit A 16x32 layout: lanes0-15 K0-7/16-23, lanes16-31 K8-15/24-31)
        Frag a0, a1;
        const unsigned short* ar0 = &Aslab[lm     ][kt * 32 + lhi * 8];
        const unsigned short* ar1 = &Aslab[16 + lm][kt * 32 + lhi * 8];
        a0.u[0] = *(const uint4*)(ar0);
        a0.u[1] = *(const uint4*)(ar0 + 16);
        a1.u[0] = *(const uint4*)(ar1);
        a1.u[1] = *(const uint4*)(ar1 + 16);

        #pragma unroll
        for (int ns = 0; ns < 4; ++ns) {
            // B fragment (bf16 B 32x16: lanes0-15 K0-7/16-23, lanes16-31 K8-15/24-31, N = lane%16)
            Frag bf;
            const unsigned short* bp =
                w1t + (size_t)(nbase + ns * 16 + lm) * DPAD + kt * 32 + lhi * 8;
            bf.u[0] = *(const uint4*)(bp);
            bf.u[1] = *(const uint4*)(bp + 16);

            acc[0][ns] = __builtin_amdgcn_wmma_f32_16x16x32_bf16(
                false, a0.v, false, bf.v, (short)0, acc[0][ns], false, false);
            acc[1][ns] = __builtin_amdgcn_wmma_f32_16x16x32_bf16(
                false, a1.v, false, bf.v, (short)0, acc[1][ns], false, false);
        }
    }

    // ---- epilogue: +b1, relu, dot with w2, reduce over N via LDS atomics ----
    float bias[4], wv[4];
    #pragma unroll
    for (int ns = 0; ns < 4; ++ns) {
        const int n = nbase + ns * 16 + lm;
        bias[ns] = b1[n];
        wv[ns]   = w2[n];
    }
    #pragma unroll
    for (int ms = 0; ms < 2; ++ms) {
        float s[8];
        #pragma unroll
        for (int r = 0; r < 8; ++r) s[r] = 0.f;
        #pragma unroll
        for (int ns = 0; ns < 4; ++ns)
            #pragma unroll
            for (int r = 0; r < 8; ++r) {
                float h = acc[ms][ns][r] + bias[ns];
                h = fmaxf(h, 0.f);
                s[r] += h * wv[ns];
            }
        // D layout: VGPR r -> M = r + 8*(lane>=16); N = lane%16
        #pragma unroll
        for (int r = 0; r < 8; ++r)
            atomicAdd(&sred[ms * 16 + lhi * 8 + r], s[r]);
    }
    __syncthreads();

    if (tid < TM) {
        const int t = t0 + tid;
        if (t < T_) scores[(size_t)b * T_ + t] = sred[tid];   // bias_2 dropped: softmax-invariant
    }
}

// ---------------- kernel 3: softmax over T (per b) ----------------
__global__ void softmax_T(const float* __restrict__ scores, float* __restrict__ A) {
    __shared__ float red[8];
    const int b    = blockIdx.x;
    const int tid  = threadIdx.x;
    const int lane = tid & 31, wave = tid >> 5;

    float m = -1e30f;
    for (int t = tid; t < T_; t += 256) m = fmaxf(m, scores[(size_t)b * T_ + t]);
    #pragma unroll
    for (int off = 16; off > 0; off >>= 1) m = fmaxf(m, __shfl_xor(m, off, 32));
    if (lane == 0) red[wave] = m;
    __syncthreads();
    float mb = red[0];
    #pragma unroll
    for (int i = 1; i < 8; ++i) mb = fmaxf(mb, red[i]);
    __syncthreads();

    float s = 0.f;
    for (int t = tid; t < T_; t += 256) s += __expf(scores[(size_t)b * T_ + t] - mb);
    #pragma unroll
    for (int off = 16; off > 0; off >>= 1) s += __shfl_xor(s, off, 32);
    if (lane == 0) red[wave] = s;
    __syncthreads();
    float sum = 0.f;
    #pragma unroll
    for (int i = 0; i < 8; ++i) sum += red[i];
    const float inv = 1.0f / sum;

    for (int t = tid; t < T_; t += 256)
        A[(size_t)b * T_ + t] = __expf(scores[(size_t)b * T_ + t] - mb) * inv;
}

// ---------------- kernel 4: weighted mean / std pooling ----------------
// one wave per (b,d) row; A row staged once per block in LDS.
__launch_bounds__(256)
__global__ void pool_stats(const float* __restrict__ x, const float* __restrict__ A,
                           float* __restrict__ out /* [B][2*D] */) {
    __shared__ float Arow[T_];
    const int b   = blockIdx.y;
    const int d0  = blockIdx.x * 8;
    const int tid = threadIdx.x;
    const int lane = tid & 31, wave = tid >> 5;

    for (int t = tid; t < T_; t += 256) Arow[t] = A[(size_t)b * T_ + t];
    __syncthreads();

    const int d = d0 + wave;
    if (d < D_) {
        const float* xr = x + ((size_t)b * D_ + d) * T_;
        float e = 0.f, s = 0.f;
        for (int i = lane; i < T_ / 4; i += 32) {      // 250 float4, coalesced over T
            const float4 v = ((const float4*)xr)[i];
            const float a0 = Arow[i * 4 + 0], a1 = Arow[i * 4 + 1];
            const float a2 = Arow[i * 4 + 2], a3 = Arow[i * 4 + 3];
            e += v.x * a0 + v.y * a1 + v.z * a2 + v.w * a3;
            s += v.x * v.x * a0 + v.y * v.y * a1 + v.z * v.z * a2 + v.w * v.w * a3;
        }
        #pragma unroll
        for (int off = 16; off > 0; off >>= 1) {
            e += __shfl_xor(e, off, 32);
            s += __shfl_xor(s, off, 32);
        }
        if (lane == 0) {
            float var = s - e * e;
            if (var < 0.f) var = 0.f;
            out[(size_t)b * (2 * D_) + d]      = e;                    // E
            out[(size_t)b * (2 * D_) + D_ + d] = sqrtf(var + 1e-5f);   // std
        }
    }
}

// ---------------- host launcher ----------------
extern "C" void kernel_launch(void* const* d_in, const int* in_sizes, int n_in,
                              void* d_out, int out_size, void* d_ws, size_t ws_size,
                              hipStream_t stream) {
    (void)in_sizes; (void)n_in; (void)out_size; (void)ws_size;

    const float* x  = (const float*)d_in[0];   // [B,D,T]
    const float* w1 = (const float*)d_in[1];   // [D,DA]
    const float* b1 = (const float*)d_in[2];   // [DA]
    const float* w2 = (const float*)d_in[3];   // [DA,1]
    // d_in[4] = bias_2: constant shift, cancels in softmax -> unused

    float* outA = (float*)d_out;               // A: [B,T] (= [B,T,1])
    float* outL = outA + (size_t)B_ * T_;      // layer_out: [B, 2*D]

    unsigned short* w1t = (unsigned short*)d_ws;                         // 1.54 MB bf16
    float* scores = (float*)((char*)d_ws + (size_t)DA_ * DPAD * sizeof(unsigned short));

    prep_w1<<<(DA_ * DPAD + 255) / 256, 256, 0, stream>>>(w1, w1t);
    attn_scores<<<dim3((T_ + TM - 1) / TM, B_), 256, 0, stream>>>(x, w1t, b1, w2, scores);
    softmax_T<<<B_, 256, 0, stream>>>(scores, outA);
    pool_stats<<<dim3((D_ + 7) / 8, B_), 256, 0, stream>>>(x, outA, outL);
}
